// MultiHeadAttention_29222957482045
// MI455X (gfx1250) — matrix operand order repaired and checked by reading, not compiled, not run on hardware
//
#include <hip/hip_runtime.h>
#include <hip/hip_bf16.h>

// ---------------- constants ----------------
#define BATCH   2
#define SEQ     2048
#define EMBED   2048
#define HEADS   16
#define HDIM    128          // EMBED / HEADS
#define N3D     6144         // 3 * EMBED
#define MROWS   4096         // BATCH * SEQ
#define SCALE   0.08838834764831843f   // 1/sqrt(128)

typedef _Float16 v16h __attribute__((ext_vector_type(16)));
typedef _Float16 v8h  __attribute__((ext_vector_type(8)));
typedef float    v8f  __attribute__((ext_vector_type(8)));
typedef float    v4f  __attribute__((ext_vector_type(4)));
typedef unsigned int v4u __attribute__((ext_vector_type(4)));
typedef unsigned int v8u __attribute__((ext_vector_type(8)));

// ---------------- WMMA fragment helpers (wave32, 16x16x32 f16) ----------------
// A-matrix 16x32 layout (ISA 7.12.2): lane r (row), half h:
//   elements 0..7  = K { 8h .. 8h+7 },  elements 8..15 = K { 16+8h .. 16+8h+7 }
__device__ __forceinline__ v16h load_a16(const _Float16* __restrict__ base, int ld,
                                         int r, int h) {
  const _Float16* p = base + r * ld;
  v8h lo = *(const v8h*)(p + h * 8);
  v8h hi = *(const v8h*)(p + 16 + h * 8);
  return __builtin_shufflevector(lo, hi, 0,1,2,3,4,5,6,7,8,9,10,11,12,13,14,15);
}

// B-matrix 32x16 layout: lane r = column, half h: elements 0..7 = K { 8h .. 8h+7 }, elements 8..15 = K { 16+8h .. 16+8h+7 }.
// `base` points at row0 of the *transposed* (column-major -> row-major) storage.
__device__ __forceinline__ v16h load_b16(const _Float16* __restrict__ base, int ld,
                                         int r, int h) {
  const _Float16* p = base + r * ld;
  v8h lo = *(const v8h*)(p + h * 8);
  v8h hi = *(const v8h*)(p + 16 + h * 8);
  return __builtin_shufflevector(lo, hi, 0,1,2,3,4,5,6,7,8,9,10,11,12,13,14,15);
}

// A-fragment sourced from an fp32 LDS row (convert on the fly)
__device__ __forceinline__ v16h load_a_f32(const float* __restrict__ p, int h) {
  v16h out;
#pragma unroll
  for (int j = 0; j < 8; ++j) {
    out[j]     = (_Float16)p[h * 8 + j];
    out[j + 8] = (_Float16)p[16 + h * 8 + j];
  }
  return out;
}

__device__ __forceinline__ v8f wmma16(v16h a, v16h b, v8f c) {
  return __builtin_amdgcn_wmma_f32_16x16x32_f16(false, a, false, b,
                                                (short)0, c, false, false);
}

// ---------------- TDM: 2D tile load, global -> LDS (ISA ch. 8 D# layout) -----
__device__ __forceinline__ void tdm_load_2d_f16(const _Float16* gsrc,
                                                unsigned lds_byte_addr,
                                                unsigned tile_d0,     // elems/row
                                                unsigned tile_d1,     // rows
                                                unsigned stride_elems,
                                                unsigned pad_en,
                                                unsigned pad_interval_code,
                                                unsigned pad_amount_code) {
  unsigned long ga = (unsigned long)gsrc;
  v4u g0;
  g0[0] = 1u;                                     // count = 1 valid descriptor
  g0[1] = lds_byte_addr;                          // lds_addr
  g0[2] = (unsigned)ga;                           // global_addr[31:0]
  g0[3] = (unsigned)(ga >> 32) | (2u << 30);      // global_addr[56:32] | type=2
  const unsigned td0 = 0x7fffffffu, td1 = 0x7fffffffu;   // huge: no OOB clip
  v8u g1;
  g1[0] = (1u << 16) | (pad_en << 20) |
          (pad_interval_code << 22) | (pad_amount_code << 25);   // data_size=2B
  g1[1] = (td0 & 0xffffu) << 16;                  // tensor_dim0[15:0]
  g1[2] = (td0 >> 16) | ((td1 & 0xffffu) << 16);  // td0[31:16] | td1[15:0]
  g1[3] = (td1 >> 16) | (tile_d0 << 16);          // td1[31:16] | tile_dim0
  g1[4] = tile_d1 & 0xffffu;                      // tile_dim1 (tile_dim2 = 0)
  g1[5] = stride_elems;                           // tensor_dim0_stride[31:0]
  g1[6] = 0u;
  g1[7] = 0u;
  asm volatile("tensor_load_to_lds %0, %1" :: "s"(g0), "s"(g1) : "memory");
}

// ---------------- kernel 1: convert X to f16 (vectorized, exact grid) --------
__global__ __launch_bounds__(256) void conv_x(const float* __restrict__ X,
                                              _Float16* __restrict__ Xh) {
  const size_t base = ((size_t)blockIdx.x * 256 + threadIdx.x) * 8;
  v4f f0 = *(const v4f*)(X + base);
  v4f f1 = *(const v4f*)(X + base + 4);
  v8h o;
#pragma unroll
  for (int j = 0; j < 4; ++j) { o[j] = (_Float16)f0[j]; o[j + 4] = (_Float16)f1[j]; }
  *(v8h*)(Xh + base) = o;
}

// ---------------- kernel 2: LDS-tiled transpose W -> Wt[n][k] (f16) ----------
#define TT 64
#define TT_LD 68
__global__ __launch_bounds__(256) void conv_wt(const float* __restrict__ W,
                                               _Float16* __restrict__ Wt) {
  __shared__ _Float16 tile[TT][TT_LD];   // 8704 B
  const int n0 = blockIdx.x * TT;        // column origin in W (row in Wt)
  const int k0 = blockIdx.y * TT;        // row origin in W
  const int t  = threadIdx.x;
  {
    const int nl = t & 63, kb = t >> 6;  // 4 rows per pass
#pragma unroll
    for (int i = 0; i < 16; ++i) {
      const int kl = kb + i * 4;
      tile[nl][kl] = (_Float16)W[(size_t)(k0 + kl) * N3D + n0 + nl];  // coalesced
    }
  }
  __syncthreads();
  {
    const int kl = t & 63, nb = t >> 6;
#pragma unroll
    for (int i = 0; i < 16; ++i) {
      const int nl = nb + i * 4;
      Wt[(size_t)(n0 + nl) * EMBED + k0 + kl] = tile[nl][kl];         // coalesced
    }
  }
}

// ---------------- kernel 3: QKV GEMM (TDM-fed, LDS double-buffered) ----------
// C[4096,6144] = Xh[4096,2048] @ W[2048,6144] + bias
// Workgroup: 128(M) x 128(N); 8 waves in 4x2; wave = 32x64 (2x4 WMMA tiles).
#define KSTEP   64
#define NSTAGE  (EMBED / KSTEP)   // 32
#define LDS_LD  88                // 64 data + 24 pad halfs (176 B rows)
#define VB_LD   136               // V write-out staging row (272 B, 16B aligned)
__global__ __launch_bounds__(256) void qkv_gemm(
    const _Float16* __restrict__ Xh, const _Float16* __restrict__ Wt,
    const float* __restrict__ bias, _Float16* __restrict__ Qh,
    _Float16* __restrict__ Kh, _Float16* __restrict__ Vt) {
  __shared__ _Float16 Atile[2][128 * LDS_LD];   // 2 x 22528 B (reused for V out)
  __shared__ _Float16 Btile[2][128 * LDS_LD];   // 2 x 22528 B

  const int tid  = threadIdx.x;
  const int wave = tid >> 5, lane = tid & 31;
  const int r = lane & 15, h = lane >> 4;
  const int wr = wave & 3, wc = wave >> 2;          // 4 row-groups x 2 col-groups
  const int rowbase = blockIdx.y * 128;             // M panel origin
  const int colbase = blockIdx.x * 128;             // N panel origin

  const _Float16* Asrc = Xh + (size_t)rowbase * EMBED;   // [128][2048]
  const _Float16* Bsrc = Wt + (size_t)colbase * EMBED;   // [128][2048]

  // pad: 32 DWORDs (=64 halfs) data -> interval code 4; 12 DWORDs pad -> code 11
  if (wave == 0) {
    tdm_load_2d_f16(Asrc, (unsigned)(size_t)&Atile[0][0], KSTEP, 128, EMBED,
                    1u, 4u, 11u);
    tdm_load_2d_f16(Bsrc, (unsigned)(size_t)&Btile[0][0], KSTEP, 128, EMBED,
                    1u, 4u, 11u);
  }

  v8f acc[2][4] = {};
  for (int ks = 0; ks < NSTAGE; ++ks) {
    const int cur = ks & 1;
    if (wave == 0) {
      if (ks + 1 < NSTAGE) {
        const int knext = (ks + 1) * KSTEP;
        tdm_load_2d_f16(Asrc + knext, (unsigned)(size_t)&Atile[cur ^ 1][0],
                        KSTEP, 128, EMBED, 1u, 4u, 11u);
        tdm_load_2d_f16(Bsrc + knext, (unsigned)(size_t)&Btile[cur ^ 1][0],
                        KSTEP, 128, EMBED, 1u, 4u, 11u);
        __builtin_amdgcn_s_wait_tensorcnt(2);   // current pair complete
      } else {
        __builtin_amdgcn_s_wait_tensorcnt(0);
      }
    }
    __syncthreads();                             // tile[cur] visible to all waves

    const _Float16* Ab = &Atile[cur][0];
    const _Float16* Bb = &Btile[cur][0];
#pragma unroll
    for (int kk = 0; kk < KSTEP; kk += 32) {
      v16h af0 = load_a16(Ab + (wr * 32) * LDS_LD + kk, LDS_LD, r, h);
      v16h af1 = load_a16(Ab + (wr * 32 + 16) * LDS_LD + kk, LDS_LD, r, h);
#pragma unroll
      for (int c = 0; c < 4; ++c) {
        v16h bf = load_b16(Bb + (wc * 64 + c * 16) * LDS_LD + kk, LDS_LD, r, h);
        acc[0][c] = wmma16(af0, bf, acc[0][c]);
        acc[1][c] = wmma16(af1, bf, acc[1][c]);
      }
    }
    __syncthreads();                             // done with tile[cur]
  }

  // ---- epilogue: section (Q/K/V) is uniform per workgroup ----
  const int wrow = rowbase + wr * 32;
  const int wcol = colbase + wc * 64;
  if (colbase < 2 * EMBED) {
    // Q or K: [b][h][s][d] stores, 32B-contiguous runs per half-wave
#pragma unroll
    for (int c = 0; c < 4; ++c) {
      const int gc   = wcol + c * 16 + r;
      const float bb = bias[gc];
      const int sec  = gc >> 11;               // 0:Q 1:K
      const int col2 = gc & 2047;
      const int hh   = col2 >> 7;
      const int dd   = col2 & 127;
#pragma unroll
      for (int t = 0; t < 2; ++t) {
#pragma unroll
        for (int e = 0; e < 8; ++e) {
          const int gr = wrow + t * 16 + e + 8 * h;
          const int bi = gr >> 11;
          const int s  = gr & 2047;
          const float v = acc[t][c][e] + bb;
          const size_t qk_idx = (((size_t)(bi * HEADS + hh) * SEQ + s) << 7) + dd;
          if (sec == 0) Qh[qk_idx] = (_Float16)(v * SCALE);
          else          Kh[qk_idx] = (_Float16)v;
        }
      }
    }
  } else {
    // V: transpose the 128x128 tile through LDS, then 256B-coalesced stores
    _Float16* Vbuf = &Atile[0][0];             // 128 x VB_LD halfs (34816 B)
#pragma unroll
    for (int c = 0; c < 4; ++c) {
      const float bb = bias[wcol + c * 16 + r];
      const int dd_l = wc * 64 + c * 16 + r;   // 0..127 (head-local dim)
#pragma unroll
      for (int t = 0; t < 2; ++t) {
#pragma unroll
        for (int e = 0; e < 8; ++e) {
          const int s_l = wr * 32 + t * 16 + e + 8 * h;   // 0..127
          Vbuf[dd_l * VB_LD + s_l] = (_Float16)(acc[t][c][e] + bb);
        }
      }
    }
    __syncthreads();
    const int bi = rowbase >> 11;
    const int s0 = rowbase & 2047;
    const int hh = (colbase - 2 * EMBED) >> 7;
    _Float16* Vrow0 = Vt + ((size_t)(bi * HEADS + hh) * HDIM) * SEQ + s0;
    const int sub = tid & 15, rg = tid >> 4;
#pragma unroll
    for (int i = 0; i < 8; ++i) {
      const int dd = i * 16 + rg;
      *(v8h*)(Vrow0 + (size_t)dd * SEQ + sub * 8) =
          *(const v8h*)&Vbuf[dd * VB_LD + sub * 8];
    }
  }
}

// ---------------- kernel 4: fused attention ----------------
// One workgroup (8 waves) per (b, h, 16-query tile).
#define PROBS_LD 2052   // 2048 + 4 pad floats (8208 B rows, 16B aligned)
__global__ __launch_bounds__(256) void attn(
    const _Float16* __restrict__ Qh, const _Float16* __restrict__ Kh,
    const _Float16* __restrict__ Vt, float* __restrict__ out_attn,
    float* __restrict__ out_scores) {
  __shared__ float probs[16][PROBS_LD];
  __shared__ float red[16][16];

  const int b = blockIdx.z, hh = blockIdx.y, qt = blockIdx.x;
  const int tid  = threadIdx.x;
  const int wave = tid >> 5, lane = tid & 31;
  const int r = lane & 15, h = lane >> 4;

  const _Float16* Qb = Qh + ((size_t)(b * HEADS + hh) * SEQ + qt * 16) * HDIM;
  const _Float16* Kb = Kh + (size_t)(b * HEADS + hh) * SEQ * HDIM;
  const _Float16* Vb = Vt + (size_t)(b * HEADS + hh) * HDIM * SEQ;

  // ---- stage 1: scores = (Q*scale) @ K^T, strip of 256 keys per wave ----
  v16h qf[4];
#pragma unroll
  for (int d = 0; d < 4; ++d) qf[d] = load_a16(Qb + d * 32, HDIM, r, h);

  for (int t = 0; t < 16; ++t) {
    const int kb = wave * 256 + t * 16;
    v8f acc = {};
#pragma unroll
    for (int d = 0; d < 4; ++d) {
      v16h bf = load_b16(Kb + (size_t)kb * HDIM + d * 32, HDIM, r, h);
      acc = wmma16(qf[d], bf, acc);
    }
#pragma unroll
    for (int e = 0; e < 8; ++e) probs[e + 8 * h][kb + r] = acc[e];
  }
  __syncthreads();

  // ---- stage 2: row softmax, vectorized (16 threads/row, 32 float4 each) ----
  const int row = tid >> 4, sub = tid & 15;
  float m = -1e30f;
  for (int j = 0; j < 32; ++j) {
    v4f x = *(const v4f*)&probs[row][sub * 4 + 64 * j];
    m = fmaxf(m, fmaxf(fmaxf(x[0], x[1]), fmaxf(x[2], x[3])));
  }
  red[row][sub] = m;
  __syncthreads();
#pragma unroll
  for (int j = 0; j < 16; ++j) m = fmaxf(m, red[row][j]);
  __syncthreads();                       // all reads of red(max) done
  float ssum = 0.f;
  for (int j = 0; j < 32; ++j) {
    float* p = &probs[row][sub * 4 + 64 * j];
    v4f x = *(const v4f*)p;
#pragma unroll
    for (int q = 0; q < 4; ++q) { x[q] = __expf(x[q] - m); ssum += x[q]; }
    *(v4f*)p = x;
  }
  red[row][sub] = ssum;
  __syncthreads();
  float tot = 0.f;
#pragma unroll
  for (int j = 0; j < 16; ++j) tot += red[row][j];
  const float inv = 1.0f / tot;

  const size_t sbase =
      (((size_t)(b * HEADS + hh) * SEQ) + qt * 16 + row) * (size_t)SEQ;
  for (int j = 0; j < 32; ++j) {
    const int c = sub * 4 + 64 * j;
    float* p = &probs[row][c];
    v4f x = *(const v4f*)p;
#pragma unroll
    for (int q = 0; q < 4; ++q) x[q] *= inv;
    *(v4f*)p = x;
    *(v4f*)&out_scores[sbase + c] = x;   // global_store_b128, 256B/16 lanes
  }
  __syncthreads();

  // ---- stage 3: out = probs @ V, wave owns 16 output dims ----
  v8f oacc = {};
  for (int k = 0; k < SEQ; k += 32) {
    v16h af = load_a_f32(&probs[r][k], h);
    v16h bf = load_b16(Vb + (size_t)(wave * 16) * SEQ + k, SEQ, r, h);
    oacc = wmma16(af, bf, oacc);
  }
#pragma unroll
  for (int e = 0; e < 8; ++e) {
    const int q  = qt * 16 + e + 8 * h;
    const int dd = wave * 16 + r;
    out_attn[((size_t)(b * SEQ + q)) * EMBED + hh * HDIM + dd] = oacc[e];
  }
}

// ---------------- host side ----------------
extern "C" void kernel_launch(void* const* d_in, const int* in_sizes, int n_in,
                              void* d_out, int out_size, void* d_ws, size_t ws_size,
                              hipStream_t stream) {
  const float* X  = (const float*)d_in[0];   // [2,2048,2048]
  const float* W  = (const float*)d_in[1];   // [2048,6144]
  const float* bq = (const float*)d_in[2];   // [6144]

  // workspace layout (f16)
  _Float16* Xh = (_Float16*)d_ws;                       // 4096*2048
  _Float16* Wt = Xh + (size_t)MROWS * EMBED;            // 6144*2048 (transposed)
  _Float16* Qh = Wt + (size_t)N3D * EMBED;              // [b][h][s][d]
  _Float16* Kh = Qh + (size_t)MROWS * EMBED;            // [b][h][s][d]
  _Float16* Vt = Kh + (size_t)MROWS * EMBED;            // [b][h][d][s]

  float* out_attn   = (float*)d_out;
  float* out_scores = out_attn + (size_t)BATCH * SEQ * EMBED;

  conv_x <<<dim3(MROWS * EMBED / (256 * 8)), 256, 0, stream>>>(X, Xh);
  conv_wt<<<dim3(N3D / TT, EMBED / TT), 256, 0, stream>>>(W, Wt);
  qkv_gemm<<<dim3(N3D / 128, MROWS / 128), 256, 0, stream>>>(Xh, Wt, bq,
                                                             Qh, Kh, Vt);
  attn<<<dim3(SEQ / 16, HEADS, BATCH), 256, 0, stream>>>(Qh, Kh, Vt,
                                                         out_attn, out_scores);
}
